// DAGMMnet_39522289057878
// MI455X (gfx1250) — matrix operand order repaired and checked by reading, not compiled, not run on hardware
//
#include <hip/hip_runtime.h>

#define NROWS 262144
#define KCOMP 16
#define DZ 8

typedef __attribute__((ext_vector_type(16))) _Float16 v16h;
typedef __attribute__((ext_vector_type(8)))  _Float16 v8h;
typedef __attribute__((ext_vector_type(8)))  float    v8f;

__device__ __forceinline__ float fast_tanh(float v) {
  float c = fminf(fmaxf(v, -10.0f), 10.0f);   // tanh saturates well before |10|
  float e = __expf(2.0f * c);
  return (e - 1.0f) / (e + 1.0f);             // branchless
}

// ---------------- conversion / init kernels ----------------

__global__ __launch_bounds__(256) void cvt_f32_f16(const float* __restrict__ s,
                                                   _Float16* __restrict__ d, size_t n) {
  size_t i = (size_t)blockIdx.x * 256 + threadIdx.x;
  if (i < n) d[i] = (_Float16)s[i];
}

// Transposed + padded weight conversion: dst[F x K], dst[c*K + r] = src[r][c] (zero pad)
__global__ __launch_bounds__(256) void cvt_pad_t(const float* __restrict__ s,
                                                 _Float16* __restrict__ d,
                                                 int Krows, int Fcols, int srows, int scols) {
  int i = blockIdx.x * 256 + threadIdx.x;
  if (i >= Krows * Fcols) return;
  int c = i / Krows, r = i - c * Krows;
  float v = (r < srows && c < scols) ? s[r * scols + c] : 0.0f;
  d[i] = (_Float16)v;
}

__global__ __launch_bounds__(256) void zero_f32(float* p, int n) {
  int i = blockIdx.x * 256 + threadIdx.x;
  if (i < n) p[i] = 0.0f;
}

// ---------------- WMMA GEMM with fused epilogue ----------------
// C[N x F] = epilogue(A[N x KD] @ W[KD x F] + bias), W transposed as Bt[F x KD].
// Each wave computes 16 rows x (16*NC) cols: one A-fragment load feeds NC WMMAs.
// All NC B-fragments are staged into registers BEFORE the WMMAs so the loads
// issue as one clause and WMMAs retire against staggered loadcnt waits.
// MODE 0: tanh -> f16 out (stride F)
// MODE 1: linear -> f16 zpad out (stride 32, cols 16..31 zeroed) + f32 z (cols<8, stride 8)
// MODE 2: softmax over the 16 cols -> f32 gamma out (stride 16)   [NC must be 1]
template<int MODE, int KD, int NC>
__global__ __launch_bounds__(256) void gemm_wmma(const _Float16* __restrict__ A,
                                                 const _Float16* __restrict__ Bt,
                                                 const float* __restrict__ bias,
                                                 int F,
                                                 _Float16* __restrict__ outF16,
                                                 float* __restrict__ outF32) {
  const int tid   = threadIdx.x;
  const int wave  = tid >> 5;
  const int lane  = tid & 31;
  const int lhalf = lane >> 4;   // 0: lanes 0-15, 1: lanes 16-31
  const int lmod  = lane & 15;
  const int rowBase = (blockIdx.x * 8 + wave) * 16;
  const int colBase = blockIdx.y * (16 * NC);

  v8f acc[NC];
#pragma unroll
  for (int c = 0; c < NC; ++c) acc[c] = (v8f){};

  // A fragment (contiguous k order): half i = K kb+16*lhalf+i (matches B)
  const _Float16* Arow = A + (size_t)(rowBase + lmod) * KD + lhalf * 16;
  // B fragment (contiguous k order): half i = Bt[(col)*KD + kb + 16*lhalf + i]
  const _Float16* Bcol0 = Bt + (size_t)(colBase + lmod) * KD + lhalf * 16;

#pragma unroll 2
  for (int kb = 0; kb < KD; kb += 32) {
    // stage ALL operands first -> loads cluster into one clause
    v8h lo = *(const v8h*)(Arow + kb);
    v8h hi = *(const v8h*)(Arow + kb + 8);
    v16h bf[NC];
#pragma unroll
    for (int c = 0; c < NC; ++c)
      bf[c] = *(const v16h*)(Bcol0 + (size_t)c * 16 * KD + kb);  // const offset folds

    v16h a;
#pragma unroll
    for (int i = 0; i < 8; ++i) { a[i] = lo[i]; a[i + 8] = hi[i]; }

#pragma unroll
    for (int c = 0; c < NC; ++c)
      acc[c] = __builtin_amdgcn_wmma_f32_16x16x32_f16(false, a, false, bf[c],
                                                      (short)0, acc[c], false, false);
  }

#pragma unroll
  for (int c = 0; c < NC; ++c) {
    const int col = colBase + c * 16 + lmod;
    float bv;
    if (MODE == 1) bv = (col < DZ) ? bias[col] : 0.0f;
    else           bv = bias[col];

#pragma unroll
    for (int r = 0; r < 8; ++r) {
      const int row = rowBase + r + lhalf * 8;   // f32 C/D layout
      float v = acc[c][r] + bv;
      if (MODE == 0) {
        outF16[(size_t)row * F + col] = (_Float16)fast_tanh(v);
      } else if (MODE == 1) {
        outF16[(size_t)row * 32 + col] = (_Float16)v;       // z padded to 32 cols
        outF16[(size_t)row * 32 + col + 16] = (_Float16)0.0f;
        if (col < DZ) outF32[(size_t)row * DZ + col] = v;   // exact z (f32)
      } else {
        float m = v;
#pragma unroll
        for (int off = 8; off; off >>= 1) m = fmaxf(m, __shfl_xor(m, off, 16));
        float e = __expf(v - m);
        float s = e;
#pragma unroll
        for (int off = 8; off; off >>= 1) s += __shfl_xor(s, off, 16);
        outF32[(size_t)row * KCOMP + col] = e / s;
      }
    }
  }
}

// ---------------- GMM statistics ----------------

__global__ __launch_bounds__(256) void reduce_moments(const float* __restrict__ gamma,
                                                      const float* __restrict__ z,
                                                      float* __restrict__ gsum,
                                                      float* __restrict__ S1, int N) {
  const int k  = threadIdx.x & 15;
  const int ns = threadIdx.x >> 4;
  float ag = 0.0f, as1[DZ];
#pragma unroll
  for (int d = 0; d < DZ; ++d) as1[d] = 0.0f;

  for (int n = blockIdx.x * 16 + ns; n < N; n += gridDim.x * 16) {
    float g = gamma[(size_t)n * KCOMP + k];
    ag += g;
#pragma unroll
    for (int d = 0; d < DZ; ++d) as1[d] += g * z[(size_t)n * DZ + d];
  }

  __shared__ float sg[KCOMP];
  __shared__ float ss[KCOMP * DZ];
  for (int i = threadIdx.x; i < KCOMP; i += 256) sg[i] = 0.0f;
  for (int i = threadIdx.x; i < KCOMP * DZ; i += 256) ss[i] = 0.0f;
  __syncthreads();
  atomicAdd(&sg[k], ag);
#pragma unroll
  for (int d = 0; d < DZ; ++d) atomicAdd(&ss[k * DZ + d], as1[d]);
  __syncthreads();
  for (int i = threadIdx.x; i < KCOMP; i += 256) atomicAdd(&gsum[i], sg[i]);
  for (int i = threadIdx.x; i < KCOMP * DZ; i += 256) atomicAdd(&S1[i], ss[i]);
}

__global__ void compute_mu(const float* __restrict__ gsum, const float* __restrict__ S1,
                           float* __restrict__ mu) {
  int i = threadIdx.x;
  if (i < KCOMP * DZ) mu[i] = S1[i] / gsum[i >> 3];
}

__global__ __launch_bounds__(256) void reduce_sigma(const float* __restrict__ gamma,
                                                    const float* __restrict__ z,
                                                    const float* __restrict__ mu,
                                                    float* __restrict__ sigTri, int N) {
  const int k  = threadIdx.x & 15;
  const int ns = threadIdx.x >> 4;
  float muk[DZ];
#pragma unroll
  for (int d = 0; d < DZ; ++d) muk[d] = mu[k * DZ + d];
  float acc[36];
#pragma unroll
  for (int i = 0; i < 36; ++i) acc[i] = 0.0f;

  for (int n = blockIdx.x * 16 + ns; n < N; n += gridDim.x * 16) {
    float g = gamma[(size_t)n * KCOMP + k];
    float d8[DZ];
#pragma unroll
    for (int l = 0; l < DZ; ++l) d8[l] = z[(size_t)n * DZ + l] - muk[l];
    int idx = 0;
#pragma unroll
    for (int l = 0; l < DZ; ++l)
#pragma unroll
      for (int m = l; m < DZ; ++m) acc[idx++] += g * d8[l] * d8[m];
  }

  __shared__ float st[KCOMP * 36];
  for (int i = threadIdx.x; i < KCOMP * 36; i += 256) st[i] = 0.0f;
  __syncthreads();
#pragma unroll
  for (int i = 0; i < 36; ++i) atomicAdd(&st[k * 36 + i], acc[i]);
  __syncthreads();
  for (int i = threadIdx.x; i < KCOMP * 36; i += 256) atomicAdd(&sigTri[i], st[i]);
}

__global__ void chol_prep(const float* __restrict__ gsum, const float* __restrict__ sigTri,
                          float* __restrict__ siginv, float* __restrict__ kconst, int N) {
  int k = threadIdx.x;
  if (k >= KCOMP) return;
  float S[DZ][DZ];
  int idx = 0;
  float inv_gs = 1.0f / gsum[k];
  for (int l = 0; l < DZ; ++l)
    for (int m = l; m < DZ; ++m) {
      float v = sigTri[k * 36 + idx++] * inv_gs;
      S[l][m] = v; S[m][l] = v;
    }
  for (int d = 0; d < DZ; ++d) S[d][d] += 1e-6f;

  float L[DZ][DZ];
  for (int i = 0; i < DZ; ++i)
    for (int j = 0; j < DZ; ++j) L[i][j] = 0.0f;
  for (int i = 0; i < DZ; ++i)
    for (int j = 0; j <= i; ++j) {
      float s = S[i][j];
      for (int p = 0; p < j; ++p) s -= L[i][p] * L[j][p];
      if (i == j) L[i][j] = sqrtf(fmaxf(s, 1e-20f));
      else        L[i][j] = s / L[j][j];
    }
  float Li[DZ][DZ];
  for (int i = 0; i < DZ; ++i)
    for (int j = 0; j < DZ; ++j) Li[i][j] = 0.0f;
  for (int c = 0; c < DZ; ++c)
    for (int i = c; i < DZ; ++i) {
      float s = (i == c) ? 1.0f : 0.0f;
      for (int p = c; p < i; ++p) s -= L[i][p] * Li[p][c];
      Li[i][c] = s / L[i][i];
    }
  for (int l = 0; l < DZ; ++l)
    for (int m = 0; m < DZ; ++m) {
      float s = 0.0f;
      for (int p = 0; p < DZ; ++p) s += Li[p][l] * Li[p][m];
      siginv[k * 64 + l * DZ + m] = s;
    }
  float logdet = 0.0f;
  for (int d = 0; d < DZ; ++d) logdet += logf(L[d][d]);
  logdet *= 2.0f;
  float phi = gsum[k] / (float)N;
  kconst[k] = logf(phi) - 0.5f * ((float)DZ * 1.8378770664093453f + logdet);
}

__global__ __launch_bounds__(256) void energy(const float* __restrict__ z,
                                              const float* __restrict__ mu,
                                              const float* __restrict__ siginv,
                                              const float* __restrict__ kconst,
                                              float* __restrict__ prob, int N) {
  int n = blockIdx.x * 256 + threadIdx.x;
  if (n >= N) return;
  float zl[DZ];
#pragma unroll
  for (int d = 0; d < DZ; ++d) zl[d] = z[(size_t)n * DZ + d];
  float lg[KCOMP];
  float mx = -3.4e38f;
#pragma unroll 4
  for (int k = 0; k < KCOMP; ++k) {
    float d8[DZ];
#pragma unroll
    for (int l = 0; l < DZ; ++l) d8[l] = zl[l] - mu[k * DZ + l];
    float q = 0.0f;
#pragma unroll
    for (int l = 0; l < DZ; ++l) {
      float t = 0.0f;
#pragma unroll
      for (int m = 0; m < DZ; ++m) t += siginv[k * 64 + l * DZ + m] * d8[m];
      q += d8[l] * t;
    }
    lg[k] = kconst[k] - 0.5f * q;
    mx = fmaxf(mx, lg[k]);
  }
  float s = 0.0f;
#pragma unroll
  for (int k = 0; k < KCOMP; ++k) s += __expf(lg[k] - mx);
  prob[n] = -(mx + logf(s));
}

// ---------------- launcher ----------------

extern "C" void kernel_launch(void* const* d_in, const int* in_sizes, int n_in,
                              void* d_out, int out_size, void* d_ws, size_t ws_size,
                              hipStream_t stream) {
  (void)in_sizes; (void)n_in; (void)out_size; (void)ws_size;
  const int N = NROWS;
  const float* x   = (const float*)d_in[0];
  const float* W0  = (const float*)d_in[1];  const float* b0  = (const float*)d_in[2];
  const float* W1  = (const float*)d_in[3];  const float* b1  = (const float*)d_in[4];
  const float* W2  = (const float*)d_in[5];  const float* b2  = (const float*)d_in[6];
  const float* W3  = (const float*)d_in[7];  const float* b3  = (const float*)d_in[8];
  const float* E0  = (const float*)d_in[9];  const float* eb0 = (const float*)d_in[10];
  const float* E1  = (const float*)d_in[11]; const float* eb1 = (const float*)d_in[12];
  const float* E2  = (const float*)d_in[13]; const float* eb2 = (const float*)d_in[14];

  float* prob = (float*)d_out;            // [N]
  float* zOut = (float*)d_out + N;        // [N,8]

  char* ws = (char*)d_ws;
  size_t off = 0;
  auto take = [&](size_t bytes) { size_t r = off; off += (bytes + 255) & ~(size_t)255; return r; };
  _Float16* x16  = (_Float16*)(ws + take((size_t)N * 128 * 2));  // reused as g0
  _Float16* h0   = (_Float16*)(ws + take((size_t)N * 512 * 2));  // reused as gamma
  _Float16* h1   = (_Float16*)(ws + take((size_t)N * 256 * 2));  // reused as g1
  _Float16* h2   = (_Float16*)(ws + take((size_t)N * 64 * 2));
  _Float16* zpad = (_Float16*)(ws + take((size_t)N * 32 * 2));   // z padded to 32 cols
  _Float16* w0t  = (_Float16*)(ws + take((size_t)512 * 128 * 2));  // [F][K]
  _Float16* w1t  = (_Float16*)(ws + take((size_t)256 * 512 * 2));
  _Float16* w2t  = (_Float16*)(ws + take((size_t)64 * 256 * 2));
  _Float16* w3t  = (_Float16*)(ws + take((size_t)16 * 64 * 2));
  _Float16* e0t  = (_Float16*)(ws + take((size_t)128 * 32 * 2));
  _Float16* e1t  = (_Float16*)(ws + take((size_t)64 * 128 * 2));
  _Float16* e2t  = (_Float16*)(ws + take((size_t)16 * 64 * 2));
  float*    accs = (float*)(ws + take((size_t)(16 + 128 + 576) * 4));
  float*    gsum   = accs;
  float*    S1     = accs + 16;
  float*    sigTri = accs + 16 + 128;
  float*    mu     = (float*)(ws + take((size_t)128 * 4));
  float*    siginv = (float*)(ws + take((size_t)1024 * 4));
  float*    kconst = (float*)(ws + take((size_t)16 * 4));
  _Float16* g0    = x16;
  _Float16* g1    = h1;
  float*    gamma = (float*)h0;

  cvt_pad_t<<<(128 * 512 + 255) / 256, 256, 0, stream>>>(W0, w0t, 128, 512, 128, 512);
  cvt_pad_t<<<(512 * 256 + 255) / 256, 256, 0, stream>>>(W1, w1t, 512, 256, 512, 256);
  cvt_pad_t<<<(256 * 64 + 255) / 256, 256, 0, stream>>>(W2, w2t, 256, 64, 256, 64);
  cvt_pad_t<<<(64 * 16 + 255) / 256, 256, 0, stream>>>(W3, w3t, 64, 16, 64, 8);
  cvt_pad_t<<<(32 * 128 + 255) / 256, 256, 0, stream>>>(E0, e0t, 32, 128, 8, 128);
  cvt_pad_t<<<(128 * 64 + 255) / 256, 256, 0, stream>>>(E1, e1t, 128, 64, 128, 64);
  cvt_pad_t<<<(64 * 16 + 255) / 256, 256, 0, stream>>>(E2, e2t, 64, 16, 64, 16);
  zero_f32<<<3, 256, 0, stream>>>(accs, 16 + 128 + 576);

  {
    size_t n = (size_t)N * 128;
    cvt_f32_f16<<<(unsigned)((n + 255) / 256), 256, 0, stream>>>(x, x16, n);
  }

  const dim3 blk(256);
  const unsigned gx = N / 128;  // 2048 row-blocks (8 waves x 16 rows)

  // encoder  (NC = column tiles per wave; A is re-read only F/(16*NC) times)
  gemm_wmma<0, 128, 8><<<dim3(gx, 4), blk, 0, stream>>>(x16, w0t, b0, 512, h0, nullptr);
  gemm_wmma<0, 512, 8><<<dim3(gx, 2), blk, 0, stream>>>(h0, w1t, b1, 256, h1, nullptr);
  gemm_wmma<0, 256, 4><<<dim3(gx, 1), blk, 0, stream>>>(h1, w2t, b2, 64, h2, nullptr);
  gemm_wmma<1, 64, 1><<<dim3(gx, 1),  blk, 0, stream>>>(h2, w3t, b3, 16, zpad, zOut);

  // estimation net
  gemm_wmma<0, 32, 8><<<dim3(gx, 1),  blk, 0, stream>>>(zpad, e0t, eb0, 128, g0, nullptr);
  gemm_wmma<0, 128, 4><<<dim3(gx, 1), blk, 0, stream>>>(g0, e1t, eb1, 64, g1, nullptr);
  gemm_wmma<2, 64, 1><<<dim3(gx, 1),  blk, 0, stream>>>(g1, e2t, eb2, 16, nullptr, gamma);

  // GMM statistics
  reduce_moments<<<2048, 256, 0, stream>>>(gamma, zOut, gsum, S1, N);
  compute_mu<<<1, 128, 0, stream>>>(gsum, S1, mu);
  reduce_sigma<<<1024, 256, 0, stream>>>(gamma, zOut, mu, sigTri, N);
  chol_prep<<<1, 32, 0, stream>>>(gsum, sigTri, siginv, kconst, N);

  // sample energy
  energy<<<N / 256, 256, 0, stream>>>(zOut, mu, siginv, kconst, prob, N);
}
